// Residual_Spatial_Enhance_Module_33698313404756
// MI455X (gfx1250) — matrix operand order repaired and checked by reading, not compiled, not run on hardware
//
#include <hip/hip_runtime.h>

typedef __attribute__((ext_vector_type(16))) _Float16 v16h;
typedef __attribute__((ext_vector_type(8)))  float    v8f;
typedef int v4i __attribute__((ext_vector_type(4)));

#define AS1 __attribute__((address_space(1)))
#define AS3 __attribute__((address_space(3)))

#define N_SP 4096
#define C_CH 256
#define CI   128
#define B_SZ 8

#if __has_builtin(__builtin_amdgcn_global_load_async_to_lds_b128) && \
    __has_builtin(__builtin_amdgcn_s_wait_asynccnt)
#define HAVE_ASYNC_LDS 1
#else
#define HAVE_ASYNC_LDS 0
#endif

__device__ inline v8f wmma_f16(v16h a, v16h b, v8f c) {
  return __builtin_amdgcn_wmma_f32_16x16x32_f16(false, a, false, b, (short)0, c, false, false);
}

// wave32 cross-lane xor broadcast via ds_bpermute (byte index = lane*4)
__device__ inline float xor_bcast(float v, int mask) {
  int lane = (int)(threadIdx.x & 31);
  int idx = ((lane ^ mask) << 2);
  return __int_as_float(__builtin_amdgcn_ds_bpermute(idx, __float_as_int(v)));
}
__device__ inline float rmax16(float v) {
  v = fmaxf(v, xor_bcast(v, 1));
  v = fmaxf(v, xor_bcast(v, 2));
  v = fmaxf(v, xor_bcast(v, 4));
  v = fmaxf(v, xor_bcast(v, 8));
  return v;
}
__device__ inline float rsum16(float v) {
  v += xor_bcast(v, 1);
  v += xor_bcast(v, 2);
  v += xor_bcast(v, 4);
  v += xor_bcast(v, 8);
  return v;
}

// -----------------------------------------------------------------------------
// Kernel 1: projection GEMM  Y = W (OD x 256, f32) * X (256 x N, f32) + bias.
// X tile (32c x 16n) staged once per block in LDS (shared by all 8 waves).
// TRANS=1 -> Y[n][o] f16 (Q,K);  TRANS=0 -> Y[o][n] f16 (V).
// -----------------------------------------------------------------------------
template <int OD, int TRANS>
__global__ void __launch_bounds__(256)
proj_kernel(const float* __restrict__ W, const float* __restrict__ bias,
            const float* __restrict__ X, _Float16* __restrict__ Y)
{
  const int N = N_SP, Cd = C_CH;
  __shared__ __align__(16) _Float16 sX[32 * 16];   // 1 KB staged B tile

  int b    = blockIdx.z;
  int n0   = blockIdx.x * 16;
  int wave = threadIdx.x >> 5;
  int lane = threadIdx.x & 31;
  int half = lane >> 4;
  int m    = lane & 15;
  int o0   = (blockIdx.y * 8 + wave) * 16;

  const float* Xb = X + (size_t)b * Cd * N + n0;
  const float* Wr = W + (size_t)(o0 + m) * Cd + half * 16;

  v8f acc = {};
  for (int c0 = 0; c0 < Cd; c0 += 32) {
    __syncthreads();
    {   // stage X[c0..c0+31][n0..n0+15] -> f16 LDS (coalesced over n)
      int cc = threadIdx.x >> 4;        // 0..15
      int nn = threadIdx.x & 15;
      sX[cc * 16 + nn]        = (_Float16)Xb[(size_t)(c0 + cc) * N + nn];
      sX[(cc + 16) * 16 + nn] = (_Float16)Xb[(size_t)(c0 + cc + 16) * N + nn];
    }
    __syncthreads();

    // A tile: W rows (contiguous f32 -> f16), k = half*16 + j
    v16h a;
#pragma unroll
    for (int j = 0; j < 8; ++j) a[j]     = (_Float16)Wr[c0 + j];
#pragma unroll
    for (int j = 0; j < 8; ++j) a[8 + j] = (_Float16)Wr[c0 + 8 + j];
    // B tile from LDS: column m, k = half*16 + j
    v16h bm;
    const _Float16* xp = sX + (size_t)half * 16 * 16 + m;
#pragma unroll
    for (int j = 0; j < 16; ++j) bm[j] = xp[j * 16];
    acc = wmma_f16(a, bm, acc);
  }

  if (TRANS) {
    _Float16* yb = Y + (size_t)b * N * OD + (size_t)(n0 + m) * OD + o0 + half * 8;
#pragma unroll
    for (int r = 0; r < 8; ++r) yb[r] = (_Float16)(acc[r] + bias[o0 + r + half * 8]);
  } else {
    _Float16* yb = Y + (size_t)b * OD * N + (size_t)(o0 + half * 8) * N + n0 + m;
#pragma unroll
    for (int r = 0; r < 8; ++r) yb[(size_t)r * N] = (_Float16)(acc[r] + bias[o0 + r + half * 8]);
  }
}

// -----------------------------------------------------------------------------
// Kernel 2: flash attention.  Block = 8 waves = 4 query-tiles x 2 c-halves.
// Qh,Kh: [b][n][ci] f16.  Vh: [b][c][n] f16.  Oh: [b][n][c] f16.
// K/V chunks double-buffered in LDS via GLOBAL_LOAD_ASYNC_TO_LDS_B128:
// chunk i+1 DMA overlaps chunk i WMMA; s_wait_asynccnt(5) retires only the
// current chunk (5 async ops/wave/chunk, in-order completion).
// -----------------------------------------------------------------------------
__global__ void __launch_bounds__(256)
attn_kernel(const _Float16* __restrict__ Qh, const _Float16* __restrict__ Kh,
            const _Float16* __restrict__ Vh, _Float16* __restrict__ Oh)
{
  const int N = N_SP, Ci = CI, C = C_CH;
  __shared__ __align__(16) _Float16 sK[2][32 * CI];      // 2 x 8 KB
  __shared__ __align__(16) _Float16 sV[2][C_CH * 32];    // 2 x 16 KB
  __shared__ __align__(16) _Float16 sP[8][16 * 32];      // 8 KB per-wave P scratch

  int b     = blockIdx.y;
  int wave  = threadIdx.x >> 5;
  int lane  = threadIdx.x & 31;
  int half  = lane >> 4;
  int m     = lane & 15;
  int qt    = wave >> 1;
  int chalf = wave & 1;
  int q0    = (blockIdx.x * 4 + qt) * 16;

  const _Float16* Qb = Qh + (size_t)b * N * Ci;
  const _Float16* Kb = Kh + (size_t)b * N * Ci;
  const _Float16* Vb = Vh + (size_t)b * C * N;

  // per-thread staging addresses (fixed within the key loop)
  int key = threadIdx.x >> 3;                 // 0..31
  int ki  = (threadIdx.x & 7) * 16;           // 0,16,..,112 (halfs)
  const _Float16* kg = Kb + (size_t)key * Ci + ki;
  int vc = threadIdx.x;                       // 0..255
  const _Float16* vg = Vb + (size_t)vc * N;

  auto stage = [&](int kc, int buf) {
    _Float16* kl = sK[buf] + key * CI + ki;
    _Float16* vl = sV[buf] + vc * 32;
#if HAVE_ASYNC_LDS
    __builtin_amdgcn_global_load_async_to_lds_b128(
        (AS1 v4i*)(kg + (size_t)kc * Ci), (AS3 v4i*)(kl), 0, 0);
    const _Float16* vs = vg + kc;
#pragma unroll
    for (int q = 0; q < 4; ++q)
      __builtin_amdgcn_global_load_async_to_lds_b128(
          (AS1 v4i*)(vs + q * 8), (AS3 v4i*)(vl + q * 8), 0, 0);
#else
    const uint4* ks = (const uint4*)(kg + (size_t)kc * Ci);
    uint4* kd = (uint4*)kl;
    kd[0] = ks[0]; kd[1] = ks[1];
    const uint4* vs = (const uint4*)(vg + kc);
    uint4* vd = (uint4*)vl;
    vd[0] = vs[0]; vd[1] = vs[1]; vd[2] = vs[2]; vd[3] = vs[3];
#endif
  };

  // Q tile resident in registers: 16 queries x 128 ci as 4 A-tiles
  v16h aQ[4];
#pragma unroll
  for (int t = 0; t < 4; ++t) {
    const _Float16* qp = Qb + (size_t)(q0 + m) * Ci + t * 32 + half * 16;
#pragma unroll
    for (int j = 0; j < 8; ++j) aQ[t][j]     = qp[j];
#pragma unroll
    for (int j = 0; j < 8; ++j) aQ[t][8 + j] = qp[8 + j];
  }

  v8f zero = {};
  v8f accO[8];
#pragma unroll
  for (int ct = 0; ct < 8; ++ct) accO[ct] = zero;
  float mrun[8], lrun[8];
#pragma unroll
  for (int r = 0; r < 8; ++r) { mrun[r] = -3.0e38f; lrun[r] = 0.0f; }

  stage(0, 0);   // prologue: chunk 0 -> buffer 0

  for (int it = 0; it < N / 32; ++it) {
    int k0  = it * 32;
    int cur = it & 1;
    bool has_next = (k0 + 32) < N;
    if (has_next) stage(k0 + 32, cur ^ 1);   // overlap next DMA with this chunk
#if HAVE_ASYNC_LDS
    if (has_next) __builtin_amdgcn_s_wait_asynccnt(5);  // retire current chunk only
    else          __builtin_amdgcn_s_wait_asynccnt(0);
#endif
    __syncthreads();   // current chunk visible block-wide

    const _Float16* sKc = sK[cur];
    const _Float16* sVc = sV[cur];

    // S = Q * K_chunk : two 16x16 tiles (keys 0..15, 16..31)
    v8f s0 = zero, s1 = zero;
#pragma unroll
    for (int t = 0; t < 4; ++t) {
      v16h b0, b1;
      const _Float16* kp0 = sKc + (size_t)m * CI        + t * 32 + half * 16;
      const _Float16* kp1 = sKc + (size_t)(16 + m) * CI + t * 32 + half * 16;
#pragma unroll
      for (int j = 0; j < 16; ++j) { b0[j] = kp0[j]; b1[j] = kp1[j]; }
      s0 = wmma_f16(aQ[t], b0, s0);
      s1 = wmma_f16(aQ[t], b1, s1);
    }

    // online softmax per row (row = r + 8*half, replicated across 16 lanes)
    float corr[8];
#pragma unroll
    for (int r = 0; r < 8; ++r) {
      float mx = rmax16(fmaxf(s0[r], s1[r]));
      float mnew = fmaxf(mrun[r], mx);
      float p0 = __expf(s0[r] - mnew);
      float p1 = __expf(s1[r] - mnew);
      s0[r] = p0; s1[r] = p1;
      float rs = rsum16(p0 + p1);
      corr[r] = __expf(mrun[r] - mnew);
      lrun[r] = lrun[r] * corr[r] + rs;
      mrun[r] = mnew;
    }
#pragma unroll
    for (int ct = 0; ct < 8; ++ct)
#pragma unroll
      for (int r = 0; r < 8; ++r) accO[ct][r] *= corr[r];

    // P: D-layout regs -> LDS -> A-layout regs (f16)
    _Float16* pw = sP[wave];
#pragma unroll
    for (int r = 0; r < 8; ++r) {
      int row = r + half * 8;
      pw[row * 32 + m]      = (_Float16)s0[r];
      pw[row * 32 + 16 + m] = (_Float16)s1[r];
    }
    v16h aP;
    {
      const _Float16* pr = pw + m * 32 + half * 16;
#pragma unroll
      for (int j = 0; j < 8; ++j) aP[j]     = pr[j];
#pragma unroll
      for (int j = 0; j < 8; ++j) aP[8 + j] = pr[8 + j];
    }

    // O += P * V_chunk over this wave's 128-channel half
#pragma unroll
    for (int ct = 0; ct < 8; ++ct) {
      v16h bv;
      const _Float16* vp = sVc + (size_t)(chalf * 128 + ct * 16 + m) * 32 + half * 16;
#pragma unroll
      for (int j = 0; j < 16; ++j) bv[j] = vp[j];
      accO[ct] = wmma_f16(aP, bv, accO[ct]);
    }

    __syncthreads();   // readers of 'cur' done before it is re-written (it+2)
  }

  // normalize by running sum and store O as [n][c] f16
#pragma unroll
  for (int r = 0; r < 8; ++r) {
    float inv = 1.0f / lrun[r];
    int qrow = q0 + r + half * 8;
#pragma unroll
    for (int ct = 0; ct < 8; ++ct) {
      float v = accO[ct][r] * inv;
      Oh[(size_t)b * N * C + (size_t)qrow * C + chalf * 128 + ct * 16 + m] = (_Float16)v;
    }
  }
}

// -----------------------------------------------------------------------------
// Kernel 3: output projection wo*O + bo, fused BN(eval) + ReLU + residual.
// A tile (O rows, shared by all waves) staged once per block in LDS.
// -----------------------------------------------------------------------------
__global__ void __launch_bounds__(256)
outproj_kernel(const _Float16* __restrict__ Oh, const float* __restrict__ wo,
               const float* __restrict__ bo, const float* __restrict__ bng,
               const float* __restrict__ bnb, const float* __restrict__ bnm,
               const float* __restrict__ bnv, const float* __restrict__ gamma,
               const float* __restrict__ xmain, float* __restrict__ out)
{
  const int N = N_SP, C = C_CH;
  __shared__ __align__(16) _Float16 sA[16 * 32];   // 1 KB staged A tile

  int b    = blockIdx.z;
  int n0   = blockIdx.x * 16;
  int wave = threadIdx.x >> 5;
  int lane = threadIdx.x & 31;
  int half = lane >> 4;
  int m    = lane & 15;
  int o0   = (blockIdx.y * 8 + wave) * 16;

  const _Float16* Ob = Oh + (size_t)b * N * C + (size_t)n0 * C;
  const float*    Wr = wo + (size_t)(o0 + m) * C + half * 16;

  v8f acc = {};
  for (int c0 = 0; c0 < C; c0 += 32) {
    __syncthreads();
    {   // stage O[n0..n0+15][c0..c0+31] (contiguous f16, coalesced over c)
      int row = threadIdx.x >> 4;      // 0..15
      int cc  = threadIdx.x & 15;
      sA[row * 32 + cc]      = Ob[(size_t)row * C + c0 + cc];
      sA[row * 32 + cc + 16] = Ob[(size_t)row * C + c0 + cc + 16];
    }
    __syncthreads();

    v16h a;
    const _Float16* ap = sA + m * 32 + half * 16;
#pragma unroll
    for (int j = 0; j < 8; ++j) a[j]     = ap[j];
#pragma unroll
    for (int j = 0; j < 8; ++j) a[8 + j] = ap[8 + j];
    v16h bm;
#pragma unroll
    for (int j = 0; j < 16; ++j) bm[j] = (_Float16)Wr[c0 + j];
    acc = wmma_f16(a, bm, acc);
  }

  int o = o0 + m;   // D column = output channel (fixed per lane)
  float inv   = bng[o] * rsqrtf(bnv[o] + 1e-5f);
  float shift = bnb[o] - bnm[o] * inv;
  float bia   = bo[o];
  float g     = gamma[0];
  size_t base = (size_t)b * C * N + (size_t)o * N + n0 + half * 8;
#pragma unroll
  for (int r = 0; r < 8; ++r) {
    float v = fmaxf((acc[r] + bia) * inv + shift, 0.0f);
    out[base + r] = xmain[base + r] + g * v;
  }
}

// -----------------------------------------------------------------------------
extern "C" void kernel_launch(void* const* d_in, const int* in_sizes, int n_in,
                              void* d_out, int out_size, void* d_ws, size_t ws_size,
                              hipStream_t stream) {
  const float* x_main  = (const float*)d_in[0];
  const float* x_guide = (const float*)d_in[1];
  const float* wq = (const float*)d_in[2];
  const float* bq = (const float*)d_in[3];
  const float* wk = (const float*)d_in[4];
  const float* bk = (const float*)d_in[5];
  const float* wv = (const float*)d_in[6];
  const float* bv = (const float*)d_in[7];
  const float* wo = (const float*)d_in[8];
  const float* bo = (const float*)d_in[9];
  const float* bng = (const float*)d_in[10];
  const float* bnb = (const float*)d_in[11];
  const float* bnm = (const float*)d_in[12];
  const float* bnv = (const float*)d_in[13];
  const float* gam = (const float*)d_in[14];

  // workspace layout (f16): Qh 8MB | Kh 8MB | Vh 16MB | Oh 16MB  (48MB total)
  _Float16* Qh = (_Float16*)d_ws;
  _Float16* Kh = Qh + (size_t)B_SZ * N_SP * CI;
  _Float16* Vh = Kh + (size_t)B_SZ * N_SP * CI;
  _Float16* Oh = Vh + (size_t)B_SZ * C_CH * N_SP;

  dim3 blk(256);
  proj_kernel<CI,   1><<<dim3(N_SP / 16, 1, B_SZ), blk, 0, stream>>>(wq, bq, x_main,  Qh);
  proj_kernel<CI,   1><<<dim3(N_SP / 16, 1, B_SZ), blk, 0, stream>>>(wk, bk, x_guide, Kh);
  proj_kernel<C_CH, 0><<<dim3(N_SP / 16, 2, B_SZ), blk, 0, stream>>>(wv, bv, x_guide, Vh);
  attn_kernel<<<dim3(N_SP / 64, B_SZ), blk, 0, stream>>>(Qh, Kh, Vh, Oh);
  outproj_kernel<<<dim3(N_SP / 16, 2, B_SZ), blk, 0, stream>>>(Oh, wo, bo, bng, bnb, bnm, bnv, gam, x_main, (float*)d_out);
}
